// SACRSN_v40_23536420782637
// MI455X (gfx1250) — compile-verified, matrix operand order checked by reading, not run on hardware
//
#include <hip/hip_runtime.h>

// ---------------------------------------------------------------------------
// CDNA5 / gfx1250 SACRSN forward.
// GEMMs (QKV complex-linear fused, VQ codebook scores, decoder) all share
// K=512 and run through one WMMA kernel:
//   - A panel (64x512 bf16) loaded once per block via TDM tensor_load_to_lds
//     with LDS padding (row stride 520 halves), waited on TENSORcnt.
//   - B streamed in 32-wide K chunks, double-buffered with
//     global_load_async_to_lds_b128 (ASYNCcnt), overlapped with WMMA.
//   - v_wmma_f32_16x16x32_bf16, 8 waves, 64x128 block tile, 4 acc/wave.
// ---------------------------------------------------------------------------

#ifndef USE_ASYNC_LDS
#define USE_ASYNC_LDS 1
#endif
#ifndef USE_TDM_A
#define USE_TDM_A 1
#endif

typedef __bf16 bf16;
typedef __attribute__((ext_vector_type(8)))  float    v8f;
typedef __attribute__((ext_vector_type(8)))  bf16     v8bf;
typedef __attribute__((ext_vector_type(16))) bf16     v16bf;
typedef __attribute__((ext_vector_type(4)))  unsigned u32x4;
typedef __attribute__((ext_vector_type(8)))  int      i32x8;
typedef __attribute__((ext_vector_type(4)))  int      i32x4;

constexpr int kB = 64;
constexpr int kT = 32;
constexpr int kD = 256;
constexpr int kV = 32000;
constexpr int kS = 32;
constexpr int k2D = 512;
constexpr int kMaxRec = 8;
constexpr float kHaltTh = 0.98f;
constexpr float kPonder = 0.005f;

// ------------------------------ device helpers -----------------------------

__device__ __forceinline__ float blk_sum256(float v, float* sh) {
  int t = threadIdx.x;
  sh[t] = v; __syncthreads();
  for (int s = 128; s > 0; s >>= 1) {
    if (t < s) sh[t] += sh[t + s];
    __syncthreads();
  }
  float r = sh[0]; __syncthreads();
  return r;
}

__device__ __forceinline__ float sigmoidf(float x) { return 1.f / (1.f + __expf(-x)); }

#if USE_ASYNC_LDS
// Async 16B global -> LDS. `mk` (generic pointer into the shared array) is a
// dummy operand that CAPTURES the LDS allocation so the "memory" clobber is
// known to cover it -- without it the compiler hoists/folds the LDS reads
// (observed in the round-1 disassembly: wmma consumed undef v[0:7]).
__device__ __forceinline__ void async_b128(unsigned lds_byte_off, const void* gaddr,
                                           const void* mk) {
  asm volatile("global_load_async_to_lds_b128 %0, %1, off"
               :: "v"(lds_byte_off), "v"(gaddr), "v"(mk) : "memory");
}
__device__ __forceinline__ void wait_asynccnt0(const void* mk) {
  asm volatile("s_wait_asynccnt 0x0" :: "v"(mk) : "memory");
}
#endif

__device__ __forceinline__ void lds_fence(const void* mk) {
  asm volatile("" :: "v"(mk) : "memory");
}

// ------------------------------ WMMA GEMM ----------------------------------
// C[M,N] = A[M,512](bf16) * Bt[N,512](bf16)^T + bias[N].  K fixed at 512.
// Block: 256 threads (8 waves); tile 64(M) x 128(N); K chunk 32.
// Wave w: rows (w&3)*16, cols (w>>2)*64 -> 4 x 16x16 accumulators.

constexpr int kK = 512;
constexpr int kLdA = kK + 8;                 // padded A row stride (halves)
constexpr int kAPanelH = 64 * kLdA;          // 33280 halves
constexpr int kBRowH = 40;                   // B LDS row stride (halves)
constexpr int kBBufH = 128 * kBRowH;         // 5120 halves per buffer

__global__ __launch_bounds__(256) void k_gemm(const bf16* __restrict__ A,
                                              const bf16* __restrict__ Bt,
                                              const float* __restrict__ bias,
                                              float* __restrict__ C,
                                              int M, int N) {
  __shared__ alignas(16) bf16 sm[kAPanelH + 2 * kBBufH];   // ~87 KB LDS
  const void* mk = (const void*)sm;

  const int tid  = threadIdx.x;
  const int bn   = blockIdx.x * 128;
  const int bm   = blockIdx.y * 64;
  const int wave = tid >> 5;
  const int lane = tid & 31;
  const int tm   = (wave & 3) * 16;
  const int tn   = (wave >> 2) * 64;
  const int lrow = lane & 15;
  const int khalf = lane >> 4;

  // ---------------- A panel: one-shot load into LDS ----------------
#if USE_TDM_A && defined(__has_builtin)
#if __has_builtin(__builtin_amdgcn_tensor_load_to_lds) && \
    __has_builtin(__builtin_amdgcn_s_wait_tensorcnt)
#define TDM_PATH 1
#endif
#endif
#ifdef TDM_PATH
  if (tid == 0) {
    unsigned long long ga = (unsigned long long)(A + (size_t)bm * kK);
    u32x4 g0;
    g0[0] = 1u;                                   // count=1, user descriptor
    g0[1] = 0u;                                   // lds_addr = 0 (panel base)
    g0[2] = (unsigned)(ga & 0xffffffffu);         // global_addr[31:0]
    g0[3] = (unsigned)((ga >> 32) & 0x1ffffffu) | (2u << 30);  // addr hi | type=2
    i32x8 g1;
    // data_size=2B (code 1), pad_enable, pad_interval=256 dwords (code 7),
    // pad_amount=4 dwords (code 3) -> row stride 520 halves in LDS.
    g1[0] = (int)((1u << 16) | (1u << 20) | (7u << 22) | (3u << 25));
    g1[1] = (int)(((unsigned)kK & 0xffffu) << 16);            // tensor_dim0 lo
    g1[2] = (int)(((unsigned)kK >> 16) | (64u << 16));        // dim0 hi | tensor_dim1 lo
    g1[3] = (int)(((unsigned)kK & 0xffffu) << 16);            // dim1 hi=0 | tile_dim0
    g1[4] = 64;                                               // tile_dim1 | tile_dim2=0
    g1[5] = kK;                                               // tensor_dim0_stride lo
    g1[6] = 0;                                                // stride hi | dim1_stride lo
    g1[7] = 0;
    i32x4 gz4 = {0, 0, 0, 0};
    i32x8 gz8 = {0, 0, 0, 0, 0, 0, 0, 0};
    // 6-arg toolchain variant: (g0, g1, g2, g3, g4, cpol)
    __builtin_amdgcn_tensor_load_to_lds(g0, g1, gz4, gz4, gz8, 0);
    __builtin_amdgcn_s_wait_tensorcnt((short)0);
  }
#else
  for (int q = 0; q < 16; ++q) {
    int s = tid + q * 256;                 // 4096 16B segments
    int row = s >> 6;                      // 64 segs per 512-half row
    int seg = (s & 63) * 8;
    *(v8bf*)&sm[row * kLdA + seg] = *(const v8bf*)(A + (size_t)(bm + row) * kK + seg);
  }
#endif
  __syncthreads();
  lds_fence(mk);

  // ---------------- B: double-buffered async streaming ----------------
  auto stageB = [&](int k0, int buf) {
#pragma unroll
    for (int q = 0; q < 2; ++q) {
      int s = tid + q * 256;               // 512 segments = 128 rows x 4
      int row = s >> 2;
      int seg = (s & 3) * 8;
      const bf16* src = Bt + (size_t)(bn + row) * kK + k0 + seg;
      __builtin_prefetch(src + 64, 0, 1);  // warm L2 two chunks ahead
      unsigned idx = (unsigned)(kAPanelH + buf * kBBufH + row * kBRowH + seg);
#if USE_ASYNC_LDS
      async_b128(idx * 2u, src, mk);
#else
      *(v8bf*)&sm[idx] = *(const v8bf*)src;
#endif
    }
  };

  stageB(0, 0);
#if USE_ASYNC_LDS
  wait_asynccnt0(mk);
#endif
  __syncthreads();

  v8f acc[4] = {};
  for (int i = 0; i < kK / 32; ++i) {
    int buf = i & 1;
    if (i + 1 < kK / 32) stageB((i + 1) * 32, buf ^ 1);

    // A fragment: lane = row tm+lrow; K = i*32 + {khalf*8..+8, 16+khalf*8..+8}
    const int arow = tm + lrow;
    const int kb = i * 32 + khalf * 8;
    v8bf alo = *(const v8bf*)&sm[arow * kLdA + kb];
    v8bf ahi = *(const v8bf*)&sm[arow * kLdA + kb + 16];
    v16bf af = __builtin_shufflevector(alo, ahi, 0,1,2,3,4,5,6,7,8,9,10,11,12,13,14,15);

    const bf16* sB = sm + kAPanelH + buf * kBBufH;
#pragma unroll
    for (int j = 0; j < 4; ++j) {
      int brow = tn + j * 16 + lrow;
      v8bf bl = *(const v8bf*)&sB[brow * kBRowH + khalf * 8];
      v8bf bh = *(const v8bf*)&sB[brow * kBRowH + 16 + khalf * 8];
      v16bf bfj = __builtin_shufflevector(bl, bh, 0,1,2,3,4,5,6,7,8,9,10,11,12,13,14,15);
      acc[j] = __builtin_amdgcn_wmma_f32_16x16x32_bf16(false, af, false, bfj,
                                                       (short)0, acc[j], false, false);
    }
#if USE_ASYNC_LDS
    wait_asynccnt0(mk);
#endif
    __syncthreads();
  }

  // C/D layout: VGPR r -> row tm + 8*(lane>>4) + r, col = (lane&15)
  const int col = lane & 15;
  const int rbase = tm + khalf * 8;
#pragma unroll
  for (int j = 0; j < 4; ++j) {
    int gn = bn + tn + j * 16 + col;
    float bv = bias ? bias[gn] : 0.f;
#pragma unroll
    for (int r = 0; r < 8; ++r) {
      C[(size_t)(bm + rbase + r) * N + gn] = acc[j][r] + bv;
    }
  }
}

// --------------------------- prep / conversion -----------------------------

__global__ __launch_bounds__(256) void k_zero(float* p, long n) {
  long i = (long)blockIdx.x * 256 + threadIdx.x;
  if (i < n) p[i] = 0.f;
}

// Combined transposed QKV weights: Wc[1536,512] bf16, bc[1536].
// Output columns: [qr qi kr ki vr vi] blocks of 256.
__global__ __launch_bounds__(256) void k_prep_qkvw(
    const float* Wqr, const float* bqr, const float* Wqi, const float* bqi,
    const float* Wkr, const float* bkr, const float* Wki, const float* bki,
    const float* Wvr, const float* bvr, const float* Wvi, const float* bvi,
    bf16* Wc, float* bc) {
  size_t e = (size_t)blockIdx.x * 256 + threadIdx.x;
  if (e >= (size_t)1536 * 512) return;
  int n = (int)(e >> 9), k = (int)(e & 511);
  int j = n >> 8, c = n & 255;
  const float *Wr, *Wi, *br, *bi;
  switch (j >> 1) {
    case 0:  Wr = Wqr; Wi = Wqi; br = bqr; bi = bqi; break;
    case 1:  Wr = Wkr; Wi = Wki; br = bkr; bi = bki; break;
    default: Wr = Wvr; Wi = Wvi; br = bvr; bi = bvi; break;
  }
  bool im = (j & 1) != 0;
  float v;
  if (!im) v = (k < 256) ? Wr[k * 256 + c] : -Wi[(k - 256) * 256 + c];
  else     v = (k < 256) ? Wi[k * 256 + c] :  Wr[(k - 256) * 256 + c];
  Wc[(size_t)n * 512 + k] = (bf16)v;
  if (k == 0) bc[n] = im ? (br[c] + bi[c]) : (br[c] - bi[c]);
}

__global__ __launch_bounds__(256) void k_f32_to_bf16(const float* in, bf16* out, long n) {
  long i = (long)blockIdx.x * 256 + threadIdx.x;
  if (i < n) out[i] = (bf16)in[i];
}

// dec_w is [2D, V]; transpose into [V, 2D] bf16.
__global__ __launch_bounds__(256) void k_prep_decw(const float* dw, bf16* out) {
  size_t e = (size_t)blockIdx.x * 256 + threadIdx.x;
  if (e >= (size_t)kV * k2D) return;
  int n = (int)(e >> 9), k = (int)(e & 511);
  out[(size_t)n * 512 + k] = (bf16)dw[(size_t)k * kV + n];
}

__global__ __launch_bounds__(256) void k_row_sumsq(const float* E, float* En2) {
  int n = blockIdx.x * 256 + threadIdx.x;
  if (n >= kV) return;
  const float* r = E + (size_t)n * k2D;
  float s = 0.f;
  for (int k = 0; k < k2D; ++k) s += r[k] * r[k];
  En2[n] = s;
}

__global__ __launch_bounds__(256) void k_embed(const int* x, const float* emb,
                                               float* xr, float* xi) {
  size_t e = (size_t)blockIdx.x * 256 + threadIdx.x;   // [T,B,D]
  if (e >= (size_t)kT * kB * kD) return;
  int d = (int)(e & 255);
  size_t tb = e >> 8;
  int b = (int)(tb & 63), t = (int)(tb >> 6);
  int tok = x[b * kT + t];
  xr[e] = emb[(size_t)tok * k2D + d];
  xi[e] = emb[(size_t)tok * k2D + kD + d];
}

// ----------------------------- recurrence ----------------------------------

__global__ __launch_bounds__(256) void k_tmix(const float* xr, const float* xi,
                                              const float* input_gate,
                                              float* gr, float* gi,
                                              float* pen, float* cum, int* stopped,
                                              int t) {
  int b = blockIdx.x, d = threadIdx.x;
  float a = sigmoidf(input_gate[0]);
  size_t o = ((size_t)t * kB + b) * kD + d;
  int g = b * kD + d;
  gr[g] = a * gr[g] + (1.f - a) * xr[o];
  gi[g] = a * gi[g] + (1.f - a) * xi[o];
  if (d == 0) { pen[b] = 0.f; cum[b] = 0.f; if (b == 0) stopped[0] = 0; }
}

// LayerNorm both channels + pack zf = [nmr nmi] as bf16 GEMM operand.
__global__ __launch_bounds__(256) void k_ln_pack(const float* gr, const float* gi,
                                                 const float* nrg, const float* nrb,
                                                 const float* nig, const float* nib,
                                                 float* nmr, float* nmi, bf16* Azf) {
  __shared__ float sh[256];
  int b = blockIdx.x, d = threadIdx.x;
  float x = gr[b * kD + d];
  float mu = blk_sum256(x, sh) * (1.f / kD);
  float xm = x - mu;
  float var = blk_sum256(xm * xm, sh) * (1.f / kD);
  float yr = xm * rsqrtf(var + 1e-5f) * nrg[d] + nrb[d];
  nmr[b * kD + d] = yr;
  Azf[(size_t)b * k2D + d] = (bf16)yr;

  float z = gi[b * kD + d];
  float mu2 = blk_sum256(z, sh) * (1.f / kD);
  float zm = z - mu2;
  float var2 = blk_sum256(zm * zm, sh) * (1.f / kD);
  float yi = zm * rsqrtf(var2 + 1e-5f) * nig[d] + nib[d];
  nmi[b * kD + d] = yi;
  Azf[(size_t)b * k2D + kD + d] = (bf16)yi;
}

// Arbitration gates: softmax over 3 logits per row. 192 threads = (b,j).
__global__ __launch_bounds__(192) void k_arb(const float* nmr, const float* nmi,
                                             const float* W, const float* bia,
                                             float* gates) {
  __shared__ float lg[192];
  int t = threadIdx.x;
  int b = t / 3, j = t % 3;
  const float* zr = nmr + b * kD;
  const float* zi = nmi + b * kD;
  float a = bia[j];
  for (int k = 0; k < kD; ++k) a += zr[k] * W[k * 3 + j];
  for (int k = 0; k < kD; ++k) a += zi[k] * W[(kD + k) * 3 + j];
  lg[t] = a; __syncthreads();
  float m = fmaxf(lg[b * 3], fmaxf(lg[b * 3 + 1], lg[b * 3 + 2]));
  float s = __expf(lg[b * 3] - m) + __expf(lg[b * 3 + 1] - m) + __expf(lg[b * 3 + 2] - m);
  gates[t] = __expf(a - m) / s;
}

// Memory read: hermitian-real similarity over S=32 slots, softmax, weighted read.
__global__ __launch_bounds__(256) void k_mem_attn(const float* mr, const float* mi,
                                                  const float* nmr, const float* nmi,
                                                  float* rr, float* ri) {
  __shared__ float part[256];
  __shared__ float attn[kS];
  int b = blockIdx.x, t = threadIdx.x;
  int s = t >> 3, p = t & 7;
  const float* mrb = mr + ((size_t)b * kS + s) * kD;
  const float* mib = mi + ((size_t)b * kS + s) * kD;
  const float* zr = nmr + b * kD;
  const float* zi = nmi + b * kD;
  float acc = 0.f;
  for (int d = p * 32; d < p * 32 + 32; ++d) acc += mrb[d] * zr[d] + mib[d] * zi[d];
  part[t] = acc; __syncthreads();
  if (p == 0) {
    float sim = 0.f;
    for (int j = 0; j < 8; ++j) sim += part[s * 8 + j];
    attn[s] = sim;
  }
  __syncthreads();
  if (t == 0) {
    float mx = attn[0];
    for (int u = 1; u < kS; ++u) mx = fmaxf(mx, attn[u]);
    float sum = 0.f;
    for (int u = 0; u < kS; ++u) { attn[u] = __expf(attn[u] - mx); sum += attn[u]; }
    for (int u = 0; u < kS; ++u) attn[u] /= sum;
  }
  __syncthreads();
  int d = t;
  float ar = 0.f, ai = 0.f;
  for (int u = 0; u < kS; ++u) {
    float w = attn[u];
    ar += w * mr[((size_t)b * kS + u) * kD + d];
    ai += w * mi[((size_t)b * kS + u) * kD + d];
  }
  rr[b * kD + d] = ar;
  ri[b * kD + d] = ai;
}

// argmin over V of (|E_n|^2 - 2 * score) -- argmin-equivalent to full d2.
__global__ __launch_bounds__(256) void k_argmin(const float* scores, const float* En2,
                                                int* idx) {
  __shared__ float sv[256];
  __shared__ int   si[256];
  int b = blockIdx.x, t = threadIdx.x;
  float best = 3.4e38f; int bi = 0;
  for (int n = t; n < kV; n += 256) {
    float v = En2[n] - 2.f * scores[(size_t)b * kV + n];
    if (v < best || (v == best && n < bi)) { best = v; bi = n; }
  }
  sv[t] = best; si[t] = bi; __syncthreads();
  for (int s = 128; s > 0; s >>= 1) {
    if (t < s && (sv[t + s] < sv[t] || (sv[t + s] == sv[t] && si[t + s] < si[t]))) {
      sv[t] = sv[t + s]; si[t] = si[t + s];
    }
    __syncthreads();
  }
  if (t == 0) idx[b] = si[0];
}

// Fused: self-gate, VQ loss, arbitration mix, 0.6/0.4 update, ACT halting.
// Writes predicated on the global `stopped` flag (== jnp.where semantics).
__global__ __launch_bounds__(256) void k_update(const float* qkv, const float* nmr,
                                                const float* nmi, const float* mrr,
                                                const float* mri, const int* idx,
                                                const float* E, const float* gates3,
                                                const float* halt_bias,
                                                float* gr, float* gi,
                                                float* pen, float* cum,
                                                const int* stopped) {
  __shared__ float sh[256];
  int b = blockIdx.x, d = threadIdx.x;
  const float* q = qkv + (size_t)b * 1536;
  float qr = q[d], qi = q[256 + d], kr = q[512 + d], ki = q[768 + d];
  float vr = q[1024 + d], vi = q[1280 + d];
  float gate = sigmoidf(blk_sum256(qr * kr + qi * ki, sh));

  int id = idx[b];
  const float* e = E + (size_t)id * k2D;
  float zr = nmr[b * kD + d], zi = nmi[b * kD + d];
  float vre = e[d], vie = e[kD + d];
  float dr = vre - zr, di = vie - zi;
  float vloss = 1.25f * blk_sum256(dr * dr + di * di, sh) / (float)k2D;

  float g0 = gates3[b * 3], g1 = gates3[b * 3 + 1], g2 = gates3[b * 3 + 2];
  float ur = g0 * (vr * gate) + g1 * mrr[b * kD + d] + g2 * vre;
  float ui = g0 * (vi * gate) + g1 * mri[b * kD + d] + g2 * vie;
  if (!stopped[0]) {
    gr[b * kD + d] = 0.6f * zr + 0.4f * ur;
    gi[b * kD + d] = 0.6f * zi + 0.4f * ui;
    if (d == 0) {
      float hb = halt_bias[0];
      float hbs = (hb > 20.f) ? hb : log1pf(__expf(hb));  // softplus
      float halt = sigmoidf(hbs - vloss);
      float c = cum[b], still = 1.f - c;
      pen[b] += still * kPonder;
      cum[b] = c + still * halt;
    }
  }
}

__global__ void k_haltflag(const float* cum, int* stopped) {
  __shared__ float sh[64];
  int t = threadIdx.x;
  sh[t] = cum[t]; __syncthreads();
  for (int s = 32; s > 0; s >>= 1) { if (t < s) sh[t] += sh[t + s]; __syncthreads(); }
  if (t == 0 && sh[0] * (1.f / kB) > kHaltTh) stopped[0] = 1;
}

// Timestep end: pack zf (for t-level VQ) + decoder row, write-gated memory roll.
__global__ __launch_bounds__(256) void k_finalize_t(const float* gr, const float* gi,
                                                    const float* srcR, const float* srcI,
                                                    float* dstR, float* dstI,
                                                    bf16* Azf, bf16* A2, int t) {
  __shared__ float sh[256];
  int b = blockIdx.x, d = threadIdx.x;
  float r = gr[b * kD + d], i = gi[b * kD + d];
  float nrm = blk_sum256(r * r + i * i, sh);
  float wg = sigmoidf(sqrtf(nrm));

  Azf[(size_t)b * k2D + d]      = (bf16)r;
  Azf[(size_t)b * k2D + kD + d] = (bf16)i;
  size_t row = (size_t)b * kT + t;      // logits row order [B,T]
  A2[row * k2D + d]      = (bf16)r;
  A2[row * k2D + kD + d] = (bf16)i;

  const float* sR = srcR + (size_t)b * kS * kD;
  const float* sI = srcI + (size_t)b * kS * kD;
  float* dR = dstR + (size_t)b * kS * kD;
  float* dI = dstI + (size_t)b * kS * kD;
  dR[d] = wg * r + (1.f - wg) * sR[(kS - 1) * kD + d];
  dI[d] = wg * i + (1.f - wg) * sI[(kS - 1) * kD + d];
  for (int s = 1; s < kS; ++s) {
    dR[s * kD + d] = sR[(s - 1) * kD + d];
    dI[s * kD + d] = sI[(s - 1) * kD + d];
  }
}

// Per-timestep scalar stats: vq loss mean, ponder mean, codebook perplexity.
__global__ void k_vq_stats(const float* gr, const float* gi, const int* idx,
                           const float* E, const float* pen, float* accs) {
  __shared__ float sh[64];
  __shared__ int sidx[64];
  int b = threadIdx.x;
  sidx[b] = idx[b];
  const float* e = E + (size_t)idx[b] * k2D;
  float s = 0.f;
  for (int d = 0; d < kD; ++d) {
    float dr = e[d] - gr[b * kD + d];
    float di = e[kD + d] - gi[b * kD + d];
    s += dr * dr + di * di;
  }
  sh[b] = 1.25f * s / (float)k2D; __syncthreads();
  for (int st = 32; st > 0; st >>= 1) { if (b < st) sh[b] += sh[b + st]; __syncthreads(); }
  if (b == 0) accs[0] += sh[0] * (1.f / kB);
  __syncthreads();
  sh[b] = pen[b]; __syncthreads();
  for (int st = 32; st > 0; st >>= 1) { if (b < st) sh[b] += sh[b + st]; __syncthreads(); }
  if (b == 0) accs[1] += sh[0] * (1.f / kB);
  __syncthreads();
  // sum_n avg_n*log(avg_n+eps) == sum_b (1/64)*log(c_{idx_b}/64 + eps)
  int my = sidx[b], c = 0;
  for (int b2 = 0; b2 < kB; ++b2) c += (sidx[b2] == my) ? 1 : 0;
  sh[b] = logf((float)c * (1.f / kB) + 1e-10f) * (1.f / kB); __syncthreads();
  for (int st = 32; st > 0; st >>= 1) { if (b < st) sh[b] += sh[b + st]; __syncthreads(); }
  if (b == 0) accs[2] += __expf(-sh[0]);
}

__global__ void k_scalars(const float* accs, float* out_tail) {
  out_tail[0] = accs[0] * (1.f / kT);
  out_tail[1] = accs[1] * (1.f / kT);
  out_tail[2] = accs[2] * (1.f / kT);
}

// ------------------------------- host side ---------------------------------

extern "C" void kernel_launch(void* const* d_in, const int* in_sizes, int n_in,
                              void* d_out, int out_size, void* d_ws, size_t ws_size,
                              hipStream_t stream) {
  (void)in_sizes; (void)n_in; (void)out_size; (void)ws_size;

  const int*   x_seq   = (const int*)  d_in[0];
  const float* enc_emb = (const float*)d_in[1];
  const float* nr_g    = (const float*)d_in[2];
  const float* nr_b    = (const float*)d_in[3];
  const float* ni_g    = (const float*)d_in[4];
  const float* ni_b    = (const float*)d_in[5];
  const float* Wqr     = (const float*)d_in[6];
  const float* bqr     = (const float*)d_in[7];
  const float* Wqi     = (const float*)d_in[8];
  const float* bqi     = (const float*)d_in[9];
  const float* Wkr     = (const float*)d_in[10];
  const float* bkr     = (const float*)d_in[11];
  const float* Wki     = (const float*)d_in[12];
  const float* bki     = (const float*)d_in[13];
  const float* Wvr     = (const float*)d_in[14];
  const float* bvr     = (const float*)d_in[15];
  const float* Wvi     = (const float*)d_in[16];
  const float* bvi     = (const float*)d_in[17];
  const float* vq_emb  = (const float*)d_in[18];
  const float* arb_w   = (const float*)d_in[19];
  const float* arb_b   = (const float*)d_in[20];
  const float* dec_w   = (const float*)d_in[21];
  const float* dec_b   = (const float*)d_in[22];
  const float* halt_b  = (const float*)d_in[23];
  const float* in_gate = (const float*)d_in[24];

  float* logits = (float*)d_out;                 // [B*T, V]

  char* wp = (char*)d_ws;
  auto alloc = [&](size_t bytes) -> void* {
    void* r = (void*)wp;
    wp += (bytes + 255) & ~(size_t)255;
    return r;
  };
  const size_t TBD = (size_t)kT * kB * kD;
  float* xr   = (float*)alloc(TBD * 4);
  float* xi   = (float*)alloc(TBD * 4);
  float* gr   = (float*)alloc((size_t)kB * kD * 4);
  float* gi   = (float*)alloc((size_t)kB * kD * 4);
  float* m0r  = (float*)alloc((size_t)kB * kS * kD * 4);
  float* m0i  = (float*)alloc((size_t)kB * kS * kD * 4);
  float* m1r  = (float*)alloc((size_t)kB * kS * kD * 4);
  float* m1i  = (float*)alloc((size_t)kB * kS * kD * 4);
  float* nmr  = (float*)alloc((size_t)kB * kD * 4);
  float* nmi  = (float*)alloc((size_t)kB * kD * 4);
  float* qkv  = (float*)alloc((size_t)kB * 1536 * 4);
  float* scores = (float*)alloc((size_t)kB * kV * 4);
  float* mrr  = (float*)alloc((size_t)kB * kD * 4);
  float* mri  = (float*)alloc((size_t)kB * kD * 4);
  float* gates3 = (float*)alloc((size_t)kB * 3 * 4);
  float* bcomb  = (float*)alloc(1536 * 4);
  float* En2    = (float*)alloc((size_t)kV * 4);
  float* pen    = (float*)alloc(kB * 4);
  float* cum    = (float*)alloc(kB * 4);
  float* accs   = (float*)alloc(4 * 4);
  int*   idx    = (int*)alloc(kB * 4);
  int*   stopped = (int*)alloc(4);
  bf16*  Azf  = (bf16*)alloc((size_t)kB * k2D * 2);
  bf16*  A2   = (bf16*)alloc((size_t)kB * kT * k2D * 2);
  bf16*  Wc   = (bf16*)alloc((size_t)1536 * k2D * 2);
  bf16*  Ebf  = (bf16*)alloc((size_t)kV * k2D * 2);
  bf16*  DecT = (bf16*)alloc((size_t)kV * k2D * 2);

  auto zero = [&](float* p, long n) {
    k_zero<<<(unsigned)((n + 255) / 256), 256, 0, stream>>>(p, n);
  };
  zero(gr, kB * kD); zero(gi, kB * kD);
  zero(m0r, (long)kB * kS * kD); zero(m0i, (long)kB * kS * kD);
  zero(m1r, (long)kB * kS * kD); zero(m1i, (long)kB * kS * kD);
  zero(accs, 4);

  k_prep_qkvw<<<(1536 * 512 + 255) / 256, 256, 0, stream>>>(
      Wqr, bqr, Wqi, bqi, Wkr, bkr, Wki, bki, Wvr, bvr, Wvi, bvi, Wc, bcomb);
  const long NE = (long)kV * k2D;
  k_f32_to_bf16<<<(unsigned)((NE + 255) / 256), 256, 0, stream>>>(vq_emb, Ebf, NE);
  k_prep_decw<<<(unsigned)((NE + 255) / 256), 256, 0, stream>>>(dec_w, DecT);
  k_row_sumsq<<<(kV + 255) / 256, 256, 0, stream>>>(vq_emb, En2);
  k_embed<<<(unsigned)((TBD + 255) / 256), 256, 0, stream>>>(x_seq, enc_emb, xr, xi);

  for (int t = 0; t < kT; ++t) {
    float* curR = (t & 1) ? m1r : m0r;
    float* curI = (t & 1) ? m1i : m0i;
    float* dstR = (t & 1) ? m0r : m1r;
    float* dstI = (t & 1) ? m0i : m1i;

    k_tmix<<<kB, kD, 0, stream>>>(xr, xi, in_gate, gr, gi, pen, cum, stopped, t);

    for (int it = 0; it < kMaxRec; ++it) {
      k_ln_pack<<<kB, kD, 0, stream>>>(gr, gi, nr_g, nr_b, ni_g, ni_b, nmr, nmi, Azf);
      k_gemm<<<dim3(1536 / 128, 1), 256, 0, stream>>>(Azf, Wc, bcomb, qkv, kB, 1536);
      k_gemm<<<dim3(kV / 128, 1), 256, 0, stream>>>(Azf, Ebf, nullptr, scores, kB, kV);
      k_arb<<<1, 192, 0, stream>>>(nmr, nmi, arb_w, arb_b, gates3);
      k_mem_attn<<<kB, kD, 0, stream>>>(curR, curI, nmr, nmi, mrr, mri);
      k_argmin<<<kB, 256, 0, stream>>>(scores, En2, idx);
      k_update<<<kB, kD, 0, stream>>>(qkv, nmr, nmi, mrr, mri, idx, vq_emb, gates3,
                                      halt_b, gr, gi, pen, cum, stopped);
      k_haltflag<<<1, 64, 0, stream>>>(cum, stopped);
    }

    k_finalize_t<<<kB, kD, 0, stream>>>(gr, gi, curR, curI, dstR, dstI, Azf, A2, t);
    k_gemm<<<dim3(kV / 128, 1), 256, 0, stream>>>(Azf, Ebf, nullptr, scores, kB, kV);
    k_argmin<<<kB, 256, 0, stream>>>(scores, En2, idx);
    k_vq_stats<<<1, 64, 0, stream>>>(gr, gi, idx, vq_emb, pen, accs);
  }

  // decoder: [B*T, 512] x [512, V]
  k_gemm<<<dim3(kV / 128, (kB * kT) / 64), 256, 0, stream>>>(
      A2, DecT, dec_b, logits, kB * kT, kV);
  k_scalars<<<1, 1, 0, stream>>>(accs, logits + (size_t)kB * kT * kV);
}
